// MultiboxLoss_9534827397456
// MI455X (gfx1250) — compile-verified, matrix operand order checked by reading, not run on hardware
//
#include <hip/hip_runtime.h>
#include <stdint.h>

#define N_PRIORS  16384
#define N_CLASSES 21
#define TILE_J    2048            // gt boxes staged per LDS tile (32 KB of float4)
#define CNT_BLOCK 256             // 8 wave32 waves
#define ROWS_PER_BLOCK 128        // 8 waves * 16 rows
#define NB_SL1  64
#define NB_CONF 64

typedef __attribute__((ext_vector_type(8))) int v8i;

// ---------------------------------------------------------------------------
// ws layout: [0] u32 positive count | +64B: float sl1_partials[64] | float conf_partials[64]
// ---------------------------------------------------------------------------

__global__ void mb_init_kernel(unsigned int* __restrict__ count) {
    if (threadIdx.x == 0) *count = 0u;
}

// ---------------------------------------------------------------------------
// Pairwise positives count: 16384 x 16384 overlap tests.
// Each wave owns 16 rows; inner loop packs 0/1 masks into the IU8 A-matrix
// (16x64 per chunk) and accumulates row counts with V_WMMA_I32_16X16X64_IU8
// against an all-ones B. Sum(all D elements) = 16 * count for the wave.
// ---------------------------------------------------------------------------
__global__ __launch_bounds__(CNT_BLOCK) void mb_count_kernel(
    const float* __restrict__ gt_locs,
    const float* __restrict__ pred_locs,
    unsigned int* __restrict__ count)
{
    __shared__ float4 gtile[TILE_J];

    const int tid    = threadIdx.x;
    const int lane   = tid & 31;
    const int wave   = tid >> 5;
    const int laneHi = (lane >> 4) & 1;
    const int row    = blockIdx.x * ROWS_PER_BLOCK + wave * 16 + (lane & 15);

    // Per-row pred bounds (note reference's swapped pairing):
    // x-overlap uses (LEFT=p0, RIGHT=p1+p3); y-overlap uses (TOP=p1, BOTTOM=p0+p2)
    const float4 p   = ((const float4*)pred_locs)[row];
    const float LEFT   = p.x;
    const float TOP    = p.y;
    const float BOTTOM = p.x + p.z;
    const float RIGHT  = p.y + p.w;

    v8i acc = {};          // i32 16x16 C/D accumulator (8 VGPRs)
    v8i bOnes;
    #pragma unroll
    for (int k = 0; k < 8; ++k) bOnes[k] = 0x01010101;  // B = 64x16 of u8 ones

    for (int j0 = 0; j0 < N_PRIORS; j0 += TILE_J) {
        // Stage gt tile: precompute (gl, gr, gt, gb) per j
        #pragma unroll
        for (int t = tid; t < TILE_J; t += CNT_BLOCK) {
            const float4 g = ((const float4*)gt_locs)[j0 + t];
            float4 s;
            s.x = g.x;         // gt_left
            s.y = g.y + g.w;   // gt_right  = gt_top  + gh
            s.z = g.y;         // gt_top
            s.w = g.x + g.z;   // gt_bottom = gt_left + gw
            gtile[t] = s;
        }
        __syncthreads();

        for (int jc = 0; jc < TILE_J; jc += 64) {
            v8i a;
            #pragma unroll
            for (int v = 0; v < 8; ++v) {
                // Documented 8-bit A-matrix 16x64 layout: K base per VGPR
                const int kbase = ((v >> 1) << 4) + ((v & 1) << 2) + (laneHi << 3);
                unsigned int word = 0u;
                #pragma unroll
                for (int b = 0; b < 4; ++b) {
                    const float4 g = gtile[jc + kbase + b];
                    float xo = fminf(RIGHT,  g.y) - fmaxf(LEFT, g.x);
                    float yo = fminf(BOTTOM, g.w) - fmaxf(TOP,  g.z);
                    xo = fmaxf(xo, 0.0f);
                    yo = fmaxf(yo, 0.0f);
                    word |= (xo * yo > 0.5f ? 1u : 0u) << (8 * b);
                }
                a[v] = (int)word;
            }
            // D = A(16x64 u8 masks) x B(ones) + C  -> per-row chunk popcounts
            acc = __builtin_amdgcn_wmma_i32_16x16x64_iu8(
                false, a, false, bOnes, acc, false, false);
        }
        __syncthreads();
    }

    // Sum all 256 D elements of this wave, divide by 16 (column replication).
    int s = 0;
    #pragma unroll
    for (int k = 0; k < 8; ++k) s += acc[k];
    #pragma unroll
    for (int off = 16; off >= 1; off >>= 1)
        s += __shfl_xor(s, off, 32);
    if (lane == 0)
        atomicAdd(count, (unsigned int)(s >> 4));   // integer: deterministic
}

// ---------------------------------------------------------------------------
// Smooth-L1 sum over 65536 elements -> fixed-order block partials
// ---------------------------------------------------------------------------
__global__ __launch_bounds__(256) void mb_sl1_kernel(
    const float* __restrict__ pred, const float* __restrict__ gt,
    float* __restrict__ partials)
{
    __shared__ float red[256];
    const int tid = threadIdx.x;
    float s = 0.0f;
    for (int i = blockIdx.x * 256 + tid; i < N_PRIORS * 4; i += NB_SL1 * 256) {
        const float d = pred[i] - gt[i];
        const float a = fabsf(d);
        s += (a < 1.0f) ? 0.5f * d * d : (a - 0.5f);
    }
    red[tid] = s;
    __syncthreads();
    #pragma unroll
    for (int off = 128; off >= 1; off >>= 1) {
        if (tid < off) red[tid] += red[tid + off];
        __syncthreads();
    }
    if (tid == 0) partials[blockIdx.x] = red[0];
}

// ---------------------------------------------------------------------------
// -log_softmax gather: one row (21 classes) per thread -> block partials
// ---------------------------------------------------------------------------
__global__ __launch_bounds__(256) void mb_conf_kernel(
    const float* __restrict__ confs, const int* __restrict__ cls,
    float* __restrict__ partials)
{
    __shared__ float red[256];
    const int tid = threadIdx.x;
    const int i   = blockIdx.x * 256 + tid;      // 64*256 == 16384 exactly
    const float* row = confs + (long)i * N_CLASSES;

    float m = row[0];
    #pragma unroll
    for (int c = 1; c < N_CLASSES; ++c) m = fmaxf(m, row[c]);
    float sum = 0.0f;
    #pragma unroll
    for (int c = 0; c < N_CLASSES; ++c) sum += expf(row[c] - m);
    const float lse = m + logf(sum);
    red[tid] = lse - row[cls[i]];                // == -(logit - lse)
    __syncthreads();
    #pragma unroll
    for (int off = 128; off >= 1; off >>= 1) {
        if (tid < off) red[tid] += red[tid + off];
        __syncthreads();
    }
    if (tid == 0) partials[blockIdx.x] = red[0];
}

// ---------------------------------------------------------------------------
// Final combine: fixed-order partial sums -> (loss_loc, loss_conf)
// ---------------------------------------------------------------------------
__global__ void mb_final_kernel(const unsigned int* __restrict__ count,
                                const float* __restrict__ sl1_partials,
                                const float* __restrict__ conf_partials,
                                float* __restrict__ out)
{
    if (threadIdx.x == 0) {
        float s1 = 0.0f;
        for (int i = 0; i < NB_SL1; ++i)  s1 += sl1_partials[i];
        float s2 = 0.0f;
        for (int i = 0; i < NB_CONF; ++i) s2 += conf_partials[i];
        const float np = (float)(*count);
        out[0] = (s1 / (float)(N_PRIORS * 4)) / np;   // loss_loc
        out[1] = s2 / (float)N_PRIORS;                // loss_conf
    }
}

extern "C" void kernel_launch(void* const* d_in, const int* in_sizes, int n_in,
                              void* d_out, int out_size, void* d_ws, size_t ws_size,
                              hipStream_t stream) {
    const float* gt_locs    = (const float*)d_in[0];
    const int*   gt_class   = (const int*)  d_in[1];
    const float* pred_locs  = (const float*)d_in[2];
    const float* pred_confs = (const float*)d_in[3];
    float* out = (float*)d_out;

    unsigned int* count   = (unsigned int*)d_ws;
    float* sl1_partials   = (float*)((char*)d_ws + 64);
    float* conf_partials  = sl1_partials + NB_SL1;

    mb_init_kernel<<<1, 32, 0, stream>>>(count);
    mb_count_kernel<<<N_PRIORS / ROWS_PER_BLOCK, CNT_BLOCK, 0, stream>>>(
        gt_locs, pred_locs, count);
    mb_sl1_kernel<<<NB_SL1, 256, 0, stream>>>(pred_locs, gt_locs, sl1_partials);
    mb_conf_kernel<<<NB_CONF, 256, 0, stream>>>(pred_confs, gt_class, conf_partials);
    mb_final_kernel<<<1, 32, 0, stream>>>(count, sl1_partials, conf_partials, out);
}
